// TransformerBlock_83047487635863
// MI455X (gfx1250) — matrix operand order repaired and checked by reading, not compiled, not run on hardware
//
#include <hip/hip_runtime.h>
#include <hip/hip_bf16.h>
#include <math.h>
#include <stdint.h>

typedef __bf16 bf16;
typedef __attribute__((ext_vector_type(16))) __bf16 v16bf;
typedef __attribute__((ext_vector_type(8)))  __bf16 bf16x8;
typedef __attribute__((ext_vector_type(8)))  float  v8f;
typedef __attribute__((ext_vector_type(4)))  int    v4i;

#define B_   2
#define S_   2048
#define D_   1024
#define H_   16
#define HD_  64
#define FF_  4096
#define T_   256
#define EPS_ 1e-5f
#define RATE_ 0.01f

// ---------------------------------------------------------------------------
// CDNA5 async global->LDS copy (ASYNCcnt-tracked), guarded so either toolchain
// compiles. The builtin takes (int4 AS1*, int4 AS3*, imm offset, imm cpol) —
// signature confirmed by the round-2 diagnostic. Falls back to VGPR round-trip
// staging when the builtin is absent.
// ---------------------------------------------------------------------------
#if defined(__HIP_DEVICE_COMPILE__) && __has_builtin(__builtin_amdgcn_global_load_async_to_lds_b128)
#define HAVE_ASYNC 1
typedef __attribute__((address_space(1))) v4i gv4i;
typedef __attribute__((address_space(3))) v4i sv4i;
__device__ inline void async_b128(const void* g, void* l) {
    __builtin_amdgcn_global_load_async_to_lds_b128(
        (gv4i*)(uintptr_t)g,
        (sv4i*)(uint32_t)(uintptr_t)l,   // generic LDS addr truncates to DS offset
        0, 0);
}
#define ASYNC_WAIT() asm volatile("s_wait_asynccnt 0" ::: "memory")
#else
#define HAVE_ASYNC 0
#define ASYNC_WAIT()
#endif

__device__ inline v8f zero8() {
    v8f z;
#pragma unroll
    for (int i = 0; i < 8; ++i) z[i] = 0.0f;
    return z;
}

// Build a 16-element bf16 fragment from two 16-byte LDS/global chunks.
__device__ inline v16bf ld16(const bf16* p0, const bf16* p1) {
    union { v16bf v; bf16x8 h[2]; } u;
    u.h[0] = *(const bf16x8*)p0;
    u.h[1] = *(const bf16x8*)p1;
    return u.v;
}

__device__ inline float gelu_exact(float v) {
    return 0.5f * v * (1.0f + erff(v * 0.70710678118654752440f));
}

// ---------------------------------------------------------------------------
// fp32 -> bf16 elementwise convert (weights)
// ---------------------------------------------------------------------------
__global__ __launch_bounds__(256) void f2bf_kernel(const float* __restrict__ in,
                                                   bf16* __restrict__ out, int n) {
    int i = blockIdx.x * 256 + threadIdx.x;
    if (i < n) out[i] = (bf16)in[i];
}

// ---------------------------------------------------------------------------
// time MLP: hidden = gelu(t @ tp1_w + tp1_b); temb = hidden @ tp2_w + tp2_b
// Tiny (2x1024 outputs) -> plain fp32 dot products.
// ---------------------------------------------------------------------------
__global__ __launch_bounds__(256) void time1_kernel(const float* __restrict__ t,
                                                    const float* __restrict__ w,
                                                    const float* __restrict__ b,
                                                    float* __restrict__ hid) {
    int j = blockIdx.x * 256 + threadIdx.x;     // [0, B_*D_)
    int bb = j >> 10, col = j & 1023;
    float acc = b[col];
    for (int k = 0; k < T_; ++k) acc += t[bb * T_ + k] * w[k * D_ + col];
    hid[j] = gelu_exact(acc);
}

__global__ __launch_bounds__(256) void time2_kernel(const float* __restrict__ hid,
                                                    const float* __restrict__ w,
                                                    const float* __restrict__ b,
                                                    float* __restrict__ temb) {
    int j = blockIdx.x * 256 + threadIdx.x;
    int bb = j >> 10, col = j & 1023;
    float acc = b[col];
    for (int k = 0; k < D_; ++k) acc += hid[bb * D_ + k] * w[k * D_ + col];
    temb[j] = acc;
}

// ---------------------------------------------------------------------------
// LayerNorm over D=1024; optional time-embedding add; bf16 output.
// One 256-thread block per row.
// ---------------------------------------------------------------------------
template <bool ADDT>
__global__ __launch_bounds__(256) void ln_kernel(const float* __restrict__ x,
                                                 const float* __restrict__ temb,
                                                 const float* __restrict__ g,
                                                 const float* __restrict__ be,
                                                 bf16* __restrict__ out) {
    __shared__ float r1[256], r2[256];
    const int row = blockIdx.x;
    const int tid = threadIdx.x;
    const int bidx = row >> 11;                 // row / S_
    float v[4];
    float s = 0.f, q = 0.f;
#pragma unroll
    for (int i = 0; i < 4; ++i) {
        int c = tid + i * 256;
        float t = x[(size_t)row * D_ + c];
        if (ADDT) t += temb[bidx * D_ + c];
        v[i] = t; s += t; q += t * t;
    }
    r1[tid] = s; r2[tid] = q;
    __syncthreads();
    for (int st = 128; st > 0; st >>= 1) {
        if (tid < st) { r1[tid] += r1[tid + st]; r2[tid] += r2[tid + st]; }
        __syncthreads();
    }
    float mean = r1[0] * (1.0f / D_);
    float var  = r2[0] * (1.0f / D_) - mean * mean;
    float rstd = rsqrtf(var + EPS_);
#pragma unroll
    for (int i = 0; i < 4; ++i) {
        int c = tid + i * 256;
        out[(size_t)row * D_ + c] = (bf16)((v[i] - mean) * rstd * g[c] + be[c]);
    }
}

// ---------------------------------------------------------------------------
// Tiled bf16 WMMA GEMM: C[M,N] = A[M,K](bf16,row) * Bw[K,N](bf16,row) + bias
// Block tile 128x128x32, 256 threads = 8 waves (4 m-groups x 2 n-groups),
// each wave computes 32x64 via 2x4 grid of v_wmma_f32_16x16x32_bf16.
// A tile staged with async global->LDS when available; next tile prefetched.
// EPI: 0 = bf16 out, 1 = GELU->bf16 out, 2 = f32 out with residual add.
// ---------------------------------------------------------------------------
template <int EPI>
__global__ __launch_bounds__(256) void gemm_bf16(const bf16* __restrict__ A,
                                                 const bf16* __restrict__ Bw,
                                                 const float* __restrict__ bias,
                                                 const float* __restrict__ res,
                                                 bf16* __restrict__ outb,
                                                 float* __restrict__ outf,
                                                 int M, int N, int K) {
    __shared__ bf16 As[128 * 40];   // [row][k] pad->40 keeps 16B-aligned frag reads
    __shared__ bf16 Bs[128 * 40];   // transposed: [n][k]
    const int tid  = threadIdx.x;
    const int lane = tid & 31;
    const int w    = tid >> 5;
    const int wm   = w & 3;         // 4 row groups of 32
    const int wn   = w >> 2;        // 2 col groups of 64
    const int rowBase = blockIdx.y * 128;
    const int colBase = blockIdx.x * 128;

    const int m0    = lane & 15;
    const int kb0   = (lane < 16) ? 0 : 8;   // A-frag k base (16-bit layout)
    const int kbb   = (lane < 16) ? 0 : 16;  // unused (B-frag reads use kb0)
    const int half8 = (lane >> 4) << 3;      // C-frag row offset

    v8f acc[2][4];
#pragma unroll
    for (int mi = 0; mi < 2; ++mi)
#pragma unroll
        for (int ni = 0; ni < 4; ++ni) acc[mi][ni] = zero8();

    for (int k0 = 0; k0 < K; k0 += 32) {
        // --- stage A (128x32, row major) and B (32x128 -> transposed) ---
#pragma unroll
        for (int s = 0; s < 2; ++s) {
            int seg = tid + s * 256;                 // 512 segments of 8 elems
            int ar = seg >> 2, ak = (seg & 3) << 3;
            const bf16* ga = A + (size_t)(rowBase + ar) * K + k0 + ak;
            bf16* la = As + ar * 40 + ak;
#if HAVE_ASYNC
            async_b128(ga, la);                      // global_load_async_to_lds_b128
#else
            *(bf16x8*)la = *(const bf16x8*)ga;
#endif
            int bk = seg >> 4, bn = (seg & 15) << 3;
            bf16x8 bv = *(const bf16x8*)(Bw + (size_t)(k0 + bk) * N + colBase + bn);
#pragma unroll
            for (int j = 0; j < 8; ++j) Bs[(bn + j) * 40 + bk] = bv[j];
        }
        // prefetch next k-step tiles into cache while this step computes
        int k1 = k0 + 32;
        if (k1 < K) {
            int ar = tid >> 2, ak = (tid & 3) << 3;
            __builtin_prefetch(A + (size_t)(rowBase + ar) * K + k1 + ak, 0, 1);
            int bk = tid >> 4, bn = (tid & 15) << 3;
            __builtin_prefetch(Bw + (size_t)(k1 + bk) * N + colBase + bn, 0, 1);
        }
        ASYNC_WAIT();
        __syncthreads();

        v16bf af[2], bfr[4];
#pragma unroll
        for (int mi = 0; mi < 2; ++mi) {
            const bf16* p = As + (wm * 32 + mi * 16 + m0) * 40 + kb0;
            af[mi] = ld16(p, p + 16);                // k: kb0..+7, kb0+16..+23
        }
#pragma unroll
        for (int ni = 0; ni < 4; ++ni) {
            const bf16* p = Bs + (wn * 64 + ni * 16 + m0) * 40 + kb0;
            bfr[ni] = ld16(p, p + 16);               // k: kb0..+7, kb0+16..+23
        }
#pragma unroll
        for (int mi = 0; mi < 2; ++mi)
#pragma unroll
            for (int ni = 0; ni < 4; ++ni)
                acc[mi][ni] = __builtin_amdgcn_wmma_f32_16x16x32_bf16(
                    false, af[mi], false, bfr[ni], (short)0, acc[mi][ni], false, false);
        __syncthreads();
    }

    // --- epilogue ---
#pragma unroll
    for (int mi = 0; mi < 2; ++mi) {
        int row0 = rowBase + wm * 32 + mi * 16 + half8;
#pragma unroll
        for (int ni = 0; ni < 4; ++ni) {
            int col = colBase + wn * 64 + ni * 16 + m0;
            float bb = bias[col];
#pragma unroll
            for (int r = 0; r < 8; ++r) {
                size_t idx = (size_t)(row0 + r) * N + col;
                float v = acc[mi][ni][r] + bb;
                if (EPI == 0)      outb[idx] = (bf16)v;
                else if (EPI == 1) outb[idx] = (bf16)gelu_exact(v);
                else               outf[idx] = v + res[idx];
            }
        }
    }
}

// ---------------------------------------------------------------------------
// Flash attention. qkv layout: [B,S,3,H,HD] bf16. Block = 128 threads
// (4 waves); block owns (b, h, 64 q rows); each wave owns 16 q rows.
// K tiles of 64 keys streamed through LDS (K row-major via async copy,
// V transposed). decay*rel_pos_bias fused per k-tile into a 128-entry LDS
// table (function of rel only), so the inner loop does LDS reads instead of
// scattered global gathers + expf. Online softmax, O += P@V via WMMA.
// ---------------------------------------------------------------------------
__global__ __launch_bounds__(128) void attn_flash(const bf16* __restrict__ qkv,
                                                  const float* __restrict__ relb,
                                                  bf16* __restrict__ out) {
    __shared__ bf16 Kt[64 * 64];        // [key][hd]
    __shared__ bf16 Vt[64 * 64];        // [hd][key]
    __shared__ bf16 Pb[4 * 16 * 64];    // per-wave P tile [m][key]
    __shared__ float FB[128];           // fused decay*bias, indexed by rel-relMin
    const int tid  = threadIdx.x;
    const int lane = tid & 31;
    const int w    = tid >> 5;
    const int qt = blockIdx.x, h = blockIdx.y, b = blockIdx.z;
    const int m0    = lane & 15;
    const int kb0   = (lane < 16) ? 0 : 8;
    const int kbb   = (lane < 16) ? 0 : 16;
    const int half8 = (lane >> 4) << 3;

    // Q fragments (two 16x32 A-frags covering hd 0..63), scale applied later
    const int qrow = qt * 64 + w * 16 + m0;
    const bf16* qp = qkv + ((size_t)(b * S_ + qrow) * 3) * D_ + h * HD_;
    v16bf qa[2];
#pragma unroll
    for (int kh = 0; kh < 2; ++kh)
        qa[kh] = ld16(qp + kh * 32 + kb0, qp + kh * 32 + kb0 + 16);

    v8f o[4];
#pragma unroll
    for (int ht = 0; ht < 4; ++ht) o[ht] = zero8();
    float mrow[8], lrow[8];
#pragma unroll
    for (int r = 0; r < 8; ++r) { mrow[r] = -1e30f; lrow[r] = 0.f; }

    for (int kt = 0; kt < S_ / 64; ++kt) {
        const int kbase = kt * 64;

        // --- fused decay*bias table for this (q-tile, k-tile): rel spans
        //     [qt*64-kbase-63, qt*64-kbase+63]; idx = rel - relMin ---
        {
            int rel = qt * 64 - kbase - 63 + tid;   // tid in [0,128)
            rel = rel < -(S_ - 1) ? -(S_ - 1) : (rel > (S_ - 1) ? (S_ - 1) : rel);
            FB[tid] = __expf(-RATE_ * fabsf((float)rel)) * relb[(rel + S_ - 1) * H_ + h];
        }

        // --- stage K (row-major, async direct-to-LDS) and V (transposed) ---
#pragma unroll
        for (int s = 0; s < 4; ++s) {
            int seg = tid + s * 128;                 // 512 segments of 8
            int key = seg >> 3, hs = (seg & 7) << 3;
            const bf16* gk = qkv + ((size_t)(b * S_ + kbase + key) * 3 + 1) * D_ + h * HD_ + hs;
#if HAVE_ASYNC
            async_b128(gk, Kt + key * 64 + hs);
#else
            *(bf16x8*)(Kt + key * 64 + hs) = *(const bf16x8*)gk;
#endif
            bf16x8 vv =
                *(const bf16x8*)(qkv + ((size_t)(b * S_ + kbase + key) * 3 + 2) * D_ + h * HD_ + hs);
#pragma unroll
            for (int j = 0; j < 8; ++j) Vt[(hs + j) * 64 + key] = vv[j];
        }
        ASYNC_WAIT();
        __syncthreads();

        // --- scores: 4 x (16x16) tiles, 2 k-steps each ---
        float sc[4][8];
#pragma unroll
        for (int nt = 0; nt < 4; ++nt) {
            v8f c = zero8();
#pragma unroll
            for (int kh = 0; kh < 2; ++kh) {
                const bf16* kp = Kt + (nt * 16 + m0) * 64 + kh * 32 + kb0;
                v16bf kf = ld16(kp, kp + 16);
                c = __builtin_amdgcn_wmma_f32_16x16x32_bf16(
                    false, qa[kh], false, kf, (short)0, c, false, false);
            }
            int bidx = w * 16 + half8 + 63 - nt * 16 - m0;  // rel-relMin for r=0
#pragma unroll
            for (int r = 0; r < 8; ++r)
                sc[nt][r] = c[r] * 0.125f + FB[bidx + r];   // 0.125 = HD^-0.5
        }

        // --- online softmax (row data spans 16 lanes in each half-wave) ---
#pragma unroll
        for (int r = 0; r < 8; ++r) {
            float mx = fmaxf(fmaxf(sc[0][r], sc[1][r]), fmaxf(sc[2][r], sc[3][r]));
#pragma unroll
            for (int off = 8; off >= 1; off >>= 1) mx = fmaxf(mx, __shfl_xor(mx, off, 32));
            float mnew  = fmaxf(mrow[r], mx);
            float alpha = __expf(mrow[r] - mnew);
            mrow[r] = mnew;
            float rs = 0.f;
#pragma unroll
            for (int nt = 0; nt < 4; ++nt) {
                float p = __expf(sc[nt][r] - mnew);
                sc[nt][r] = p; rs += p;
            }
#pragma unroll
            for (int off = 8; off >= 1; off >>= 1) rs += __shfl_xor(rs, off, 32);
            lrow[r] = lrow[r] * alpha + rs;
#pragma unroll
            for (int ht = 0; ht < 4; ++ht) o[ht][r] *= alpha;
        }

        // --- P (C-layout) -> LDS (row-major) for A-frag reload ---
        bf16* pb = Pb + w * 1024;
#pragma unroll
        for (int nt = 0; nt < 4; ++nt)
#pragma unroll
            for (int r = 0; r < 8; ++r)
                pb[(half8 + r) * 64 + nt * 16 + m0] = (bf16)sc[nt][r];
        asm volatile("s_wait_dscnt 0" ::: "memory");   // wave-local LDS RAW

        // --- O += P(16x64) @ V(64x64) ---
#pragma unroll
        for (int kh = 0; kh < 2; ++kh) {
            const bf16* pp = pb + m0 * 64 + kh * 32 + kb0;
            v16bf pf = ld16(pp, pp + 16);
#pragma unroll
            for (int ht = 0; ht < 4; ++ht) {
                const bf16* vp = Vt + (ht * 16 + m0) * 64 + kh * 32 + kb0;
                v16bf vf = ld16(vp, vp + 16);
                o[ht] = __builtin_amdgcn_wmma_f32_16x16x32_bf16(
                    false, pf, false, vf, (short)0, o[ht], false, false);
            }
        }
        __syncthreads();
    }

    // --- normalize and write bf16 attention output [B,S,H*HD] ---
#pragma unroll
    for (int r = 0; r < 8; ++r) {
        float inv = 1.0f / lrow[r];
        int row = qt * 64 + w * 16 + half8 + r;
#pragma unroll
        for (int ht = 0; ht < 4; ++ht)
            out[(size_t)(b * S_ + row) * D_ + h * HD_ + ht * 16 + m0] =
                (bf16)(o[ht][r] * inv);
    }
}

// ---------------------------------------------------------------------------
// Launch
// ---------------------------------------------------------------------------
extern "C" void kernel_launch(void* const* d_in, const int* in_sizes, int n_in,
                              void* d_out, int out_size, void* d_ws, size_t ws_size,
                              hipStream_t stream) {
    const float* x      = (const float*)d_in[0];
    const float* t      = (const float*)d_in[1];
    const float* qkv_w  = (const float*)d_in[2];
    const float* qkv_b  = (const float*)d_in[3];
    const float* proj_w = (const float*)d_in[4];
    const float* proj_b = (const float*)d_in[5];
    const float* relb   = (const float*)d_in[6];
    const float* ln1_g  = (const float*)d_in[7];
    const float* ln1_b  = (const float*)d_in[8];
    const float* ln2_g  = (const float*)d_in[9];
    const float* ln2_b  = (const float*)d_in[10];
    const float* ff1_w  = (const float*)d_in[11];
    const float* ff1_b  = (const float*)d_in[12];
    const float* ff2_w  = (const float*)d_in[13];
    const float* ff2_b  = (const float*)d_in[14];
    const float* tp1_w  = (const float*)d_in[15];
    const float* tp1_b  = (const float*)d_in[16];
    const float* tp2_w  = (const float*)d_in[17];
    const float* tp2_b  = (const float*)d_in[18];
    float* outp = (float*)d_out;

    const int BS = B_ * S_;                         // 4096
    char* ws = (char*)d_ws;
    size_t off = 0;
    auto alloc = [&](size_t bytes) { size_t r = off; off += (bytes + 255) & ~(size_t)255; return r; };

    bf16*  xnorm = (bf16*)(ws + alloc((size_t)BS * D_ * 2));
    bf16*  qkvb  = (bf16*)(ws + alloc((size_t)BS * 3 * D_ * 2));
    bf16*  attnb = (bf16*)(ws + alloc((size_t)BS * D_ * 2));
    float* x1    = (float*)(ws + alloc((size_t)BS * D_ * 4));
    bf16*  h2    = (bf16*)(ws + alloc((size_t)BS * D_ * 2));
    bf16*  ffact = (bf16*)(ws + alloc((size_t)BS * FF_ * 2));
    bf16*  wqkv  = (bf16*)(ws + alloc((size_t)D_ * 3 * D_ * 2));
    bf16*  wproj = (bf16*)(ws + alloc((size_t)D_ * D_ * 2));
    bf16*  wff1  = (bf16*)(ws + alloc((size_t)D_ * FF_ * 2));
    bf16*  wff2  = (bf16*)(ws + alloc((size_t)FF_ * D_ * 2));
    float* thid  = (float*)(ws + alloc((size_t)B_ * D_ * 4));
    float* temb  = (float*)(ws + alloc((size_t)B_ * D_ * 4));

    // weight conversions fp32 -> bf16
    f2bf_kernel<<<(D_ * 3 * D_) / 256, 256, 0, stream>>>(qkv_w, wqkv, D_ * 3 * D_);
    f2bf_kernel<<<(D_ * D_) / 256, 256, 0, stream>>>(proj_w, wproj, D_ * D_);
    f2bf_kernel<<<(D_ * FF_) / 256, 256, 0, stream>>>(ff1_w, wff1, D_ * FF_);
    f2bf_kernel<<<(FF_ * D_) / 256, 256, 0, stream>>>(ff2_w, wff2, FF_ * D_);

    // time MLP
    time1_kernel<<<(B_ * D_) / 256, 256, 0, stream>>>(t, tp1_w, tp1_b, thid);
    time2_kernel<<<(B_ * D_) / 256, 256, 0, stream>>>(thid, tp2_w, tp2_b, temb);

    // LN1 (with time-emb add) -> bf16
    ln_kernel<true><<<BS, 256, 0, stream>>>(x, temb, ln1_g, ln1_b, xnorm);

    // QKV GEMM -> bf16 [B,S,3,H,HD]
    gemm_bf16<0><<<dim3(3 * D_ / 128, BS / 128), 256, 0, stream>>>(
        xnorm, wqkv, qkv_b, nullptr, qkvb, nullptr, BS, 3 * D_, D_);

    // flash attention -> bf16 [B,S,D]
    attn_flash<<<dim3(S_ / 64, H_, B_), 128, 0, stream>>>(qkvb, relb, attnb);

    // proj GEMM + residual(x) -> x1 (f32)
    gemm_bf16<2><<<dim3(D_ / 128, BS / 128), 256, 0, stream>>>(
        attnb, wproj, proj_b, x, nullptr, x1, BS, D_, D_);

    // LN2 -> bf16
    ln_kernel<false><<<BS, 256, 0, stream>>>(x1, nullptr, ln2_g, ln2_b, h2);

    // FF1 GEMM + exact GELU -> bf16
    gemm_bf16<1><<<dim3(FF_ / 128, BS / 128), 256, 0, stream>>>(
        h2, wff1, ff1_b, nullptr, ffact, nullptr, BS, FF_, D_);

    // FF2 GEMM + residual(x1) -> d_out (f32)
    gemm_bf16<2><<<dim3(D_ / 128, BS / 128), 256, 0, stream>>>(
        ffact, wff2, ff2_b, x1, nullptr, outp, BS, D_, FF_);
}
